// MPNNNet_59811714564607
// MI455X (gfx1250) — matrix operand order repaired and checked by reading, not compiled, not run on hardware
//
#include <hip/hip_runtime.h>

// ---------------------------------------------------------------------------
// MPNN (NNConv + GRU + Set2Set) for MI455X / gfx1250.
// Heavy GEMMs -> v_wmma_f32_16x16x32_bf16 (f32 accum).
// All WMMA B-operands stored [col][K] (transposed) so each lane's fragment is
// 2 contiguous b128 loads instead of 16 scalar u16 loads.
// We (E x 64 x 64) stored bf16 AND pre-transposed to [e][o][i] (via a one-time
// column permutation of nn_w2) so the dominant 1.5GB matvec stream is read
// with b128 loads. Per-edge matvec stays VALU: it is bandwidth-bound and every
// edge has its own B-matrix (WMMA would be 1/16-utilized).
// ---------------------------------------------------------------------------

#define DIMH   64
#define NFEAT  29
#define EFEAT  6
#define NGRAPH 512
#define NNODE  16384
#define NEDGE  32768
#define HID    128        // edge-MLP hidden width
#define WCOLS  4096       // DIMH*DIMH

typedef __attribute__((ext_vector_type(16))) __bf16 v16bf;
typedef __attribute__((ext_vector_type(8)))  float  v8f;

__device__ __forceinline__ __bf16 f2bf(float f) {
  unsigned u = __float_as_uint(f);
  u += 0x7FFFu + ((u >> 16) & 1u);             // round-to-nearest-even
  return __builtin_bit_cast(__bf16, (unsigned short)(u >> 16));
}
__device__ __forceinline__ float bf2f(__bf16 h) {
  unsigned short s = __builtin_bit_cast(unsigned short, h);
  return __uint_as_float(((unsigned)s) << 16);
}
__device__ __forceinline__ float bflo(unsigned u) { return __uint_as_float(u << 16); }
__device__ __forceinline__ float bfhi(unsigned u) { return __uint_as_float(u & 0xffff0000u); }
__device__ __forceinline__ float sigm(float x) { return 1.f / (1.f + expf(-x)); }

__device__ __forceinline__ void atomicMaxFloat(float* addr, float val) {
  int* ai = (int*)addr;
  int cur = __float_as_int(*addr);
  while (__int_as_float(cur) < val) {
    int prev = atomicCAS(ai, cur, __float_as_int(val));
    if (prev == cur) break;
    cur = prev;
  }
}

// ---------------------------------------------------------------------------
// Utility kernels
// ---------------------------------------------------------------------------
__global__ void fill_kernel(float* p, float v, int n) {
  int i = blockIdx.x * blockDim.x + threadIdx.x;
  if (i < n) p[i] = v;
}

__global__ void cvt_bf16_kernel(const float* __restrict__ src, __bf16* __restrict__ dst, int n) {
  int i = blockIdx.x * blockDim.x + threadIdx.x;
  if (i < n) dst[i] = f2bf(src[i]);
}

// src [rows, cols] -> dst [cols, rows] (bf16)
__global__ void cvt_transpose_kernel(const float* __restrict__ src, __bf16* __restrict__ dst,
                                     int rows, int cols) {
  int i = blockIdx.x * blockDim.x + threadIdx.x;
  if (i < rows * cols) {
    int r = i / cols, c = i % cols;
    dst[c * rows + r] = f2bf(src[i]);
  }
}

// nn_w2 [128,4096] -> w2bt [4096,128] bf16 with column remap c = o*64+i
// (original col = i*64+o).  Coalesced reads, scattered one-time stores (1MB).
__global__ void w2_permute_kernel(const float* __restrict__ w2, __bf16* __restrict__ w2bt) {
  int i = blockIdx.x * blockDim.x + threadIdx.x;       // over 128*4096
  if (i >= HID * WCOLS) return;
  int k = i / WCOLS, col0 = i % WCOLS;                 // col0 = in*64 + out
  int c = ((col0 & 63) << 6) | (col0 >> 6);            // c = out*64 + in
  w2bt[(size_t)c * HID + k] = f2bf(w2[i]);
}

// ---------------------------------------------------------------------------
// lin0: h = relu(x @ lin0_w + b)   [N,29]x[29,64]
// ---------------------------------------------------------------------------
__global__ void lin0_kernel(const float* __restrict__ x, const float* __restrict__ w,
                            const float* __restrict__ b, float* __restrict__ h) {
  __shared__ float xs[NFEAT];
  int node = blockIdx.x, d = threadIdx.x;
  if (d < NFEAT) xs[d] = x[node * NFEAT + d];
  __syncthreads();
  float acc = b[d];
  #pragma unroll
  for (int i = 0; i < NFEAT; ++i) acc += xs[i] * w[i * DIMH + d];
  h[node * DIMH + d] = fmaxf(acc, 0.f);
}

// ---------------------------------------------------------------------------
// edge MLP layer 1: hbf = bf16(relu(edge_attr @ nn_w1 + b1))  [E,6]x[6,128]
// ---------------------------------------------------------------------------
__global__ void edge_mlp1_kernel(const float* __restrict__ ea, const float* __restrict__ w1,
                                 const float* __restrict__ b1, __bf16* __restrict__ hbf) {
  __shared__ float es[EFEAT];
  int e = blockIdx.x, j = threadIdx.x;
  if (j < EFEAT) es[j] = ea[e * EFEAT + j];
  __syncthreads();
  float acc = b1[j];
  #pragma unroll
  for (int i = 0; i < EFEAT; ++i) acc += es[i] * w1[i * HID + j];
  hbf[e * HID + j] = f2bf(fmaxf(acc, 0.f));
}

// ---------------------------------------------------------------------------
// edge MLP layer 2 (the big GEMM): We[e, o*64+i] = hbf[E,128] @ w2bt + b2
// 16x16 tile per wave, K=128 in 4 WMMA steps. A: 2xb128/step, B: 2xb128/step.
// ---------------------------------------------------------------------------
__global__ void edge_w2_wmma_kernel(const __bf16* __restrict__ hbf,
                                    const __bf16* __restrict__ w2bt,
                                    const float*  __restrict__ b2,
                                    __bf16* __restrict__ We) {
  const int lane = threadIdx.x;
  const int lr = lane & 15, hi = lane >> 4;
  const int n0 = blockIdx.x * 16;       // permuted column tile (o*64+i order)
  const int m0 = blockIdx.y * 16;       // edge tile

  v8f acc;
  const int c = n0 + lr;                          // permuted col
  const int oc = ((c & 63) << 6) | (c >> 6);      // original col for bias
  float bias = b2[oc];
  #pragma unroll
  for (int v = 0; v < 8; ++v) acc[v] = bias;

  #pragma unroll
  for (int kk = 0; kk < 4; ++kk) {
    v16bf a;
    const int arow = m0 + lr;
    const int kbA = kk * 32 + hi * 16;
    #pragma unroll
    for (int j = 0; j < 8; ++j) {
      a[j]     = hbf[arow * HID + kbA + j];        // 16B contiguous
      a[j + 8] = hbf[arow * HID + kbA + 8 + j];    // 16B contiguous
    }
    v16bf b;
    const int kbB = kk * 32 + hi * 16;
    #pragma unroll
    for (int j = 0; j < 16; ++j)
      b[j] = w2bt[(size_t)c * HID + kbB + j];      // 32B contiguous
    acc = __builtin_amdgcn_wmma_f32_16x16x32_bf16(false, a, false, b, (short)0, acc, false, false);
  }
  #pragma unroll
  for (int v = 0; v < 8; ++v) {
    int row = m0 + v + 8 * hi;
    We[(size_t)row * WCOLS + c] = f2bf(acc[v]);
  }
}

// ---------------------------------------------------------------------------
// degree
// ---------------------------------------------------------------------------
__global__ void deg_count_kernel(const int* __restrict__ ei, float* __restrict__ deg) {
  int e = blockIdx.x * blockDim.x + threadIdx.x;
  if (e < NEDGE) atomicAdd(&deg[ei[NEDGE + e]], 1.f);
}

// ---------------------------------------------------------------------------
// per-edge matvec + scatter: agg[dst,o] += sum_i h[src,i] * We[e, o*64+i]
// We is [e][o][i]: thread o streams its 128B row as 8x b128 loads.
// ---------------------------------------------------------------------------
__global__ void msg_kernel(const float* __restrict__ h, const __bf16* __restrict__ We,
                           const int* __restrict__ ei, float* __restrict__ agg) {
  __shared__ float xs[DIMH];
  int e = blockIdx.x, d = threadIdx.x;
  int src = ei[e], dst = ei[NEDGE + e];
  xs[d] = h[src * DIMH + d];
  const __bf16* W = We + (size_t)e * WCOLS + d * DIMH;
  __builtin_prefetch(W, 0, 1);                 // global_prefetch_b8: stream hint
  __syncthreads();
  const uint4* Wr = (const uint4*)W;           // 8 bf16 per uint4
  float acc = 0.f;
  #pragma unroll
  for (int cidx = 0; cidx < 8; ++cidx) {
    uint4 pk = Wr[cidx];
    int i = cidx * 8;
    acc += xs[i + 0] * bflo(pk.x);
    acc += xs[i + 1] * bfhi(pk.x);
    acc += xs[i + 2] * bflo(pk.y);
    acc += xs[i + 3] * bfhi(pk.y);
    acc += xs[i + 4] * bflo(pk.z);
    acc += xs[i + 5] * bfhi(pk.z);
    acc += xs[i + 6] * bflo(pk.w);
    acc += xs[i + 7] * bfhi(pk.w);
  }
  atomicAdd(&agg[dst * DIMH + d], acc);
}

// ---------------------------------------------------------------------------
// NNConv combine: m = relu(h @ conv_root + agg/deg + bias)   WMMA, K=64,N=64
// wconvT is [o][i] so B fragments are contiguous.
// ---------------------------------------------------------------------------
__global__ void conv_wmma_kernel(const float* __restrict__ h, const __bf16* __restrict__ wconvT,
                                 const float* __restrict__ agg, const float* __restrict__ deg,
                                 const float* __restrict__ bias, float* __restrict__ m) {
  const int lane = threadIdx.x;
  const int lr = lane & 15, hi = lane >> 4;
  const int m0 = blockIdx.x * 16;
  v8f acc[4];
  #pragma unroll
  for (int nt = 0; nt < 4; ++nt)
    #pragma unroll
    for (int v = 0; v < 8; ++v) {
      int row = m0 + v + 8 * hi;
      int col = nt * 16 + lr;
      float dg = fmaxf(deg[row], 1.f);
      acc[nt][v] = agg[row * DIMH + col] / dg + bias[col];
    }
  #pragma unroll
  for (int kk = 0; kk < 2; ++kk) {
    v16bf a;
    const int arow = m0 + lr;
    const int kbA = kk * 32 + hi * 16;
    #pragma unroll
    for (int j = 0; j < 8; ++j) {
      a[j]     = f2bf(h[arow * DIMH + kbA + j]);
      a[j + 8] = f2bf(h[arow * DIMH + kbA + 8 + j]);
    }
    const int kbB = kk * 32 + hi * 16;
    #pragma unroll
    for (int nt = 0; nt < 4; ++nt) {
      v16bf b;
      #pragma unroll
      for (int j = 0; j < 16; ++j)
        b[j] = wconvT[(nt * 16 + lr) * DIMH + kbB + j];   // 32B contiguous
      acc[nt] = __builtin_amdgcn_wmma_f32_16x16x32_bf16(false, a, false, b, (short)0, acc[nt], false, false);
    }
  }
  #pragma unroll
  for (int nt = 0; nt < 4; ++nt)
    #pragma unroll
    for (int v = 0; v < 8; ++v) {
      int row = m0 + v + 8 * hi;
      m[row * DIMH + nt * 16 + lr] = fmaxf(acc[nt][v], 0.f);
    }
}

// ---------------------------------------------------------------------------
// GRU gate GEMM: Out[N,192] = A[N,64] @ W^T + bias   (WMMA)
// W is the torch [192(out),64(in)] matrix used DIRECTLY (it is already the
// transposed-B layout): B[col][k] = W[col*64+k].
// ---------------------------------------------------------------------------
__global__ void gemm192_wmma_kernel(const float* __restrict__ A, const __bf16* __restrict__ W,
                                    const float* __restrict__ bias, float* __restrict__ Out) {
  const int lane = threadIdx.x;
  const int lr = lane & 15, hi = lane >> 4;
  const int m0 = blockIdx.x * 16;
  v8f acc[12];
  #pragma unroll
  for (int nt = 0; nt < 12; ++nt) {
    float b = bias[nt * 16 + lr];
    #pragma unroll
    for (int v = 0; v < 8; ++v) acc[nt][v] = b;
  }
  #pragma unroll
  for (int kk = 0; kk < 2; ++kk) {
    v16bf a;
    const int arow = m0 + lr;
    const int kbA = kk * 32 + hi * 16;
    #pragma unroll
    for (int j = 0; j < 8; ++j) {
      a[j]     = f2bf(A[arow * DIMH + kbA + j]);
      a[j + 8] = f2bf(A[arow * DIMH + kbA + 8 + j]);
    }
    const int kbB = kk * 32 + hi * 16;
    #pragma unroll
    for (int nt = 0; nt < 12; ++nt) {
      v16bf b;
      #pragma unroll
      for (int j = 0; j < 16; ++j)
        b[j] = W[(nt * 16 + lr) * DIMH + kbB + j];        // 32B contiguous
      acc[nt] = __builtin_amdgcn_wmma_f32_16x16x32_bf16(false, a, false, b, (short)0, acc[nt], false, false);
    }
  }
  #pragma unroll
  for (int nt = 0; nt < 12; ++nt)
    #pragma unroll
    for (int v = 0; v < 8; ++v) {
      int row = m0 + v + 8 * hi;
      Out[row * 192 + nt * 16 + lr] = acc[nt][v];
    }
}

// ---------------------------------------------------------------------------
// GRU elementwise (torch gate order r,z,n):  h updated in place
// ---------------------------------------------------------------------------
__global__ void gru_elem_kernel(const float* __restrict__ GI, const float* __restrict__ GH,
                                float* __restrict__ h) {
  int n = blockIdx.x, d = threadIdx.x;
  const float* gi = GI + n * 192;
  const float* gh = GH + n * 192;
  float r  = sigm(gi[d] + gh[d]);
  float z  = sigm(gi[64 + d] + gh[64 + d]);
  float nn = tanhf(gi[128 + d] + r * gh[128 + d]);
  float hv = h[n * DIMH + d];
  h[n * DIMH + d] = (1.f - z) * nn + z * hv;
}

// ---------------------------------------------------------------------------
// Set2Set LSTM (tiny; VALU)
// ---------------------------------------------------------------------------
__global__ void lstm_gemm_kernel(const float* __restrict__ qstar, const float* __restrict__ hl,
                                 const float* __restrict__ wih, const float* __restrict__ bih,
                                 const float* __restrict__ whh, const float* __restrict__ bhh,
                                 float* __restrict__ gbuf) {
  __shared__ float qs[2 * DIMH];
  __shared__ float hs[DIMH];
  int g = blockIdx.x, c = threadIdx.x;       // 256 threads
  if (c < 128) qs[c] = qstar[g * 128 + c];
  else if (c < 192) hs[c - 128] = hl[g * DIMH + (c - 128)];
  __syncthreads();
  float acc = bih[c] + bhh[c];
  #pragma unroll 8
  for (int j = 0; j < 128; ++j) acc += qs[j] * wih[c * 128 + j];
  #pragma unroll 8
  for (int j = 0; j < 64; ++j)  acc += hs[j] * whh[c * 64 + j];
  gbuf[g * 256 + c] = acc;
}

__global__ void lstm_elem_kernel(const float* __restrict__ gbuf, float* __restrict__ hl,
                                 float* __restrict__ cl) {
  int g = blockIdx.x, d = threadIdx.x;
  const float* gr = gbuf + g * 256;
  float i  = sigm(gr[d]);
  float f  = sigm(gr[64 + d]);
  float gg = tanhf(gr[128 + d]);
  float o  = sigm(gr[192 + d]);
  float c  = f * cl[g * DIMH + d] + i * gg;
  cl[g * DIMH + d] = c;
  hl[g * DIMH + d] = o * tanhf(c);
}

// ---------------------------------------------------------------------------
// Attention (segment softmax over sorted batch)
// ---------------------------------------------------------------------------
__global__ void attn_score_kernel(const float* __restrict__ h, const float* __restrict__ hl,
                                  const int* __restrict__ batch, float* __restrict__ evec) {
  int n = blockIdx.x * blockDim.x + threadIdx.x;
  if (n >= NNODE) return;
  const float* q = hl + batch[n] * DIMH;
  const float* hv = h + n * DIMH;
  float acc = 0.f;
  #pragma unroll 8
  for (int d = 0; d < DIMH; ++d) acc += hv[d] * q[d];
  evec[n] = acc;
}

__global__ void attn_max_kernel(const float* __restrict__ evec, const int* __restrict__ batch,
                                float* __restrict__ emax) {
  int n = blockIdx.x * blockDim.x + threadIdx.x;
  if (n < NNODE) atomicMaxFloat(&emax[batch[n]], evec[n]);
}

__global__ void attn_exp_kernel(float* __restrict__ evec, const int* __restrict__ batch,
                                const float* __restrict__ emax, float* __restrict__ asum) {
  int n = blockIdx.x * blockDim.x + threadIdx.x;
  if (n >= NNODE) return;
  int b = batch[n];
  float a = expf(evec[n] - emax[b]);
  evec[n] = a;
  atomicAdd(&asum[b], a);
}

__global__ void attn_rvec_kernel(const float* __restrict__ evec, const int* __restrict__ batch,
                                 const float* __restrict__ asum, const float* __restrict__ h,
                                 float* __restrict__ rvec) {
  int n = blockIdx.x, d = threadIdx.x;
  int b = batch[n];
  float w = evec[n] / (asum[b] + 1e-16f);
  atomicAdd(&rvec[b * DIMH + d], w * h[n * DIMH + d]);
}

__global__ void qstar_kernel(const float* __restrict__ hl, const float* __restrict__ rvec,
                             float* __restrict__ qstar) {
  int i = blockIdx.x * blockDim.x + threadIdx.x;
  if (i >= NGRAPH * 128) return;
  int g = i >> 7, j = i & 127;
  qstar[i] = (j < 64) ? hl[g * DIMH + j] : rvec[g * DIMH + (j - 64)];
}

// ---------------------------------------------------------------------------
// Readout: y = relu(q_star @ lin1_w + b1) @ lin2_w + b2
// ---------------------------------------------------------------------------
__global__ void readout_kernel(const float* __restrict__ qstar, const float* __restrict__ w1,
                               const float* __restrict__ b1, const float* __restrict__ w2,
                               const float* __restrict__ b2, float* __restrict__ y) {
  __shared__ float qs[2 * DIMH];
  __shared__ float red[DIMH];
  int g = blockIdx.x, d = threadIdx.x;
  qs[d] = qstar[g * 128 + d];
  qs[d + 64] = qstar[g * 128 + d + 64];
  __syncthreads();
  float acc = b1[d];
  #pragma unroll 8
  for (int j = 0; j < 128; ++j) acc += qs[j] * w1[j * DIMH + d];
  acc = fmaxf(acc, 0.f);
  red[d] = acc * w2[d];
  __syncthreads();
  for (int s = 32; s > 0; s >>= 1) {
    if (d < s) red[d] += red[d + s];
    __syncthreads();
  }
  if (d == 0) y[g] = red[0] + b2[0];
}

// ---------------------------------------------------------------------------
// Host orchestration
// ---------------------------------------------------------------------------
extern "C" void kernel_launch(void* const* d_in, const int* in_sizes, int n_in,
                              void* d_out, int out_size, void* d_ws, size_t ws_size,
                              hipStream_t stream) {
  const float* x         = (const float*)d_in[0];
  const int*   ei        = (const int*)  d_in[1];
  const float* eattr     = (const float*)d_in[2];
  const int*   batch     = (const int*)  d_in[3];
  const float* lin0_w    = (const float*)d_in[4];
  const float* lin0_b    = (const float*)d_in[5];
  const float* nn_w1     = (const float*)d_in[6];
  const float* nn_b1     = (const float*)d_in[7];
  const float* nn_w2     = (const float*)d_in[8];
  const float* nn_b2     = (const float*)d_in[9];
  const float* conv_root = (const float*)d_in[10];
  const float* conv_bias = (const float*)d_in[11];
  const float* gru_wih   = (const float*)d_in[12];
  const float* gru_whh   = (const float*)d_in[13];
  const float* gru_bih   = (const float*)d_in[14];
  const float* gru_bhh   = (const float*)d_in[15];
  const float* lstm_wih  = (const float*)d_in[16];
  const float* lstm_whh  = (const float*)d_in[17];
  const float* lstm_bih  = (const float*)d_in[18];
  const float* lstm_bhh  = (const float*)d_in[19];
  const float* lin1_w    = (const float*)d_in[20];
  const float* lin1_b    = (const float*)d_in[21];
  const float* lin2_w    = (const float*)d_in[22];
  const float* lin2_b    = (const float*)d_in[23];
  float* y = (float*)d_out;

  char* ws = (char*)d_ws;
  size_t off = 0;
  auto carve = [&](size_t bytes) -> char* {
    char* p = ws + off;
    off += (bytes + 255) & ~(size_t)255;
    return p;
  };
  float*  h      = (float*) carve((size_t)NNODE * DIMH * 4);
  float*  m      = (float*) carve((size_t)NNODE * DIMH * 4);
  float*  agg    = (float*) carve((size_t)NNODE * DIMH * 4);
  float*  deg    = (float*) carve((size_t)NNODE * 4);
  __bf16* hbf    = (__bf16*)carve((size_t)NEDGE * HID * 2);
  __bf16* w2bt   = (__bf16*)carve((size_t)WCOLS * HID * 2);
  __bf16* We     = (__bf16*)carve((size_t)NEDGE * WCOLS * 2);
  __bf16* wconvT = (__bf16*)carve((size_t)DIMH * DIMH * 2);
  __bf16* wgih   = (__bf16*)carve((size_t)192 * DIMH * 2);
  __bf16* wghh   = (__bf16*)carve((size_t)192 * DIMH * 2);
  float*  GI     = (float*) carve((size_t)NNODE * 192 * 4);
  float*  GH     = (float*) carve((size_t)NNODE * 192 * 4);
  float*  qstar  = (float*) carve((size_t)NGRAPH * 128 * 4);
  float*  hl     = (float*) carve((size_t)NGRAPH * DIMH * 4);
  float*  cl     = (float*) carve((size_t)NGRAPH * DIMH * 4);
  float*  glstm  = (float*) carve((size_t)NGRAPH * 256 * 4);
  float*  evec   = (float*) carve((size_t)NNODE * 4);
  float*  emax   = (float*) carve((size_t)NGRAPH * 4);
  float*  asum   = (float*) carve((size_t)NGRAPH * 4);
  float*  rvec   = (float*) carve((size_t)NGRAPH * DIMH * 4);

  auto cdiv = [](int a, int b) { return (a + b - 1) / b; };

  // --- one-time prep ---
  fill_kernel<<<cdiv(NNODE, 256), 256, 0, stream>>>(deg, 0.f, NNODE);
  deg_count_kernel<<<cdiv(NEDGE, 256), 256, 0, stream>>>(ei, deg);

  lin0_kernel<<<NNODE, DIMH, 0, stream>>>(x, lin0_w, lin0_b, h);
  edge_mlp1_kernel<<<NEDGE, HID, 0, stream>>>(eattr, nn_w1, nn_b1, hbf);

  w2_permute_kernel<<<cdiv(HID * WCOLS, 256), 256, 0, stream>>>(nn_w2, w2bt);
  cvt_transpose_kernel<<<cdiv(DIMH * DIMH, 256), 256, 0, stream>>>(conv_root, wconvT, DIMH, DIMH);
  cvt_bf16_kernel<<<cdiv(192 * DIMH, 256), 256, 0, stream>>>(gru_wih, wgih, 192 * DIMH);
  cvt_bf16_kernel<<<cdiv(192 * DIMH, 256), 256, 0, stream>>>(gru_whh, wghh, 192 * DIMH);

  // big GEMM: We (transposed layout) = hbf @ w2bt + b2   (WMMA bf16)
  edge_w2_wmma_kernel<<<dim3(WCOLS / 16, NEDGE / 16), 32, 0, stream>>>(hbf, w2bt, nn_b2, We);

  // --- 6 message-passing iterations ---
  for (int it = 0; it < 6; ++it) {
    fill_kernel<<<cdiv(NNODE * DIMH, 256), 256, 0, stream>>>(agg, 0.f, NNODE * DIMH);
    msg_kernel<<<NEDGE, DIMH, 0, stream>>>(h, We, ei, agg);
    conv_wmma_kernel<<<NNODE / 16, 32, 0, stream>>>(h, wconvT, agg, deg, conv_bias, m);
    gemm192_wmma_kernel<<<NNODE / 16, 32, 0, stream>>>(m, wgih, gru_bih, GI);
    gemm192_wmma_kernel<<<NNODE / 16, 32, 0, stream>>>(h, wghh, gru_bhh, GH);
    gru_elem_kernel<<<NNODE, DIMH, 0, stream>>>(GI, GH, h);
  }

  // --- Set2Set pooling, 3 steps ---
  fill_kernel<<<cdiv(NGRAPH * 128, 256), 256, 0, stream>>>(qstar, 0.f, NGRAPH * 128);
  fill_kernel<<<cdiv(NGRAPH * DIMH, 256), 256, 0, stream>>>(hl, 0.f, NGRAPH * DIMH);
  fill_kernel<<<cdiv(NGRAPH * DIMH, 256), 256, 0, stream>>>(cl, 0.f, NGRAPH * DIMH);
  for (int s = 0; s < 3; ++s) {
    lstm_gemm_kernel<<<NGRAPH, 256, 0, stream>>>(qstar, hl, lstm_wih, lstm_bih,
                                                 lstm_whh, lstm_bhh, glstm);
    lstm_elem_kernel<<<NGRAPH, DIMH, 0, stream>>>(glstm, hl, cl);

    fill_kernel<<<cdiv(NGRAPH, 256), 256, 0, stream>>>(emax, -1e30f, NGRAPH);
    fill_kernel<<<cdiv(NGRAPH, 256), 256, 0, stream>>>(asum, 0.f, NGRAPH);
    fill_kernel<<<cdiv(NGRAPH * DIMH, 256), 256, 0, stream>>>(rvec, 0.f, NGRAPH * DIMH);

    attn_score_kernel<<<cdiv(NNODE, 256), 256, 0, stream>>>(h, hl, batch, evec);
    attn_max_kernel<<<cdiv(NNODE, 256), 256, 0, stream>>>(evec, batch, emax);
    attn_exp_kernel<<<cdiv(NNODE, 256), 256, 0, stream>>>(evec, batch, emax, asum);
    attn_rvec_kernel<<<NNODE, DIMH, 0, stream>>>(evec, batch, asum, h, rvec);
    qstar_kernel<<<cdiv(NGRAPH * 128, 256), 256, 0, stream>>>(hl, rvec, qstar);
  }

  // --- readout ---
  readout_kernel<<<NGRAPH, DIMH, 0, stream>>>(qstar, lin1_w, lin1_b, lin2_w, lin2_b, y);
}
